// std_GCN_34600256536636
// MI455X (gfx1250) — compile-verified, matrix operand order checked by reading, not run on hardware
//
#include <hip/hip_runtime.h>
#include <hip/hip_bf16.h>

typedef __attribute__((ext_vector_type(16))) _Float16 v16h;
typedef __attribute__((ext_vector_type(8)))  float    v8f;
typedef __attribute__((ext_vector_type(4)))  unsigned int v4u;

#define HF 128  // feature / hidden width

// ---------------------------------------------------------------------------
// 1) Row-normalize feat (f32 [N,128]) -> f16 [N,128].  One wave per node.
// ---------------------------------------------------------------------------
__global__ __launch_bounds__(256) void rownorm_kernel(
    const float* __restrict__ feat, _Float16* __restrict__ xh, int N) {
  int wid  = (blockIdx.x * blockDim.x + threadIdx.x) >> 5;
  int lane = threadIdx.x & 31;
  if (wid >= N) return;
  const float4* f = (const float4*)(feat + (size_t)wid * HF);
  float4 v = f[lane];
  float s = v.x + v.y + v.z + v.w;
#pragma unroll
  for (int off = 16; off; off >>= 1) s += __shfl_xor(s, off, 32);
  float inv = 1.0f / s;
  _Float16* o = xh + (size_t)wid * HF + lane * 4;
  o[0] = (_Float16)(v.x * inv);
  o[1] = (_Float16)(v.y * inv);
  o[2] = (_Float16)(v.z * inv);
  o[3] = (_Float16)(v.w * inv);
}

// ---------------------------------------------------------------------------
// 2) Degree / normalization
// ---------------------------------------------------------------------------
__global__ void deg_init_kernel(float* __restrict__ deg, int N) {
  int i = blockIdx.x * blockDim.x + threadIdx.x;
  if (i < N) deg[i] = 1.0f;  // self-loop weight
}

__global__ void deg_accum_kernel(const int* __restrict__ col,
                                 const float* __restrict__ w,
                                 float* __restrict__ deg, int E) {
  int e = blockIdx.x * blockDim.x + threadIdx.x;
  if (e < E) atomicAdd(&deg[col[e]], w[e]);
}

__global__ void dinv_kernel(const float* __restrict__ deg,
                            float* __restrict__ dinv, int N) {
  int i = blockIdx.x * blockDim.x + threadIdx.x;
  if (i < N) { float d = deg[i]; dinv[i] = d > 0.0f ? rsqrtf(d) : 0.0f; }
}

__global__ void edgenorm_kernel(const int* __restrict__ row,
                                const int* __restrict__ col,
                                const float* __restrict__ w,
                                const float* __restrict__ dinv,
                                float* __restrict__ nrm, int E) {
  int e = blockIdx.x * blockDim.x + threadIdx.x;
  if (e < E) nrm[e] = dinv[row[e]] * w[e] * dinv[col[e]];
}

// ---------------------------------------------------------------------------
// 3) Transpose W (f32 [128,128]) -> Wt f16 [n][k]
// ---------------------------------------------------------------------------
__global__ void transpose_w_kernel(const float* __restrict__ W,
                                   _Float16* __restrict__ Wt) {
  int k = blockIdx.x;    // 128
  int n = threadIdx.x;   // 128
  Wt[n * HF + k] = (_Float16)W[k * HF + n];
}

// ---------------------------------------------------------------------------
// 4) WMMA GEMM: XP[N,128] = X[N,128](f16) @ W[128,128] (via Wt, f16), f32 out.
//    Block = 256 thr = 8 waves.  The 128x128 A panel is staged in LDS once
//    (8x ds_store_b128 per thread), then each wave owns col-tile n0=16*wave
//    and sweeps 8 row tiles; K=128 -> 4x v_wmma_f32_16x16x32_f16 per tile.
//    A fragments come from LDS (2x ds_load_b128), B fragments from global.
// ---------------------------------------------------------------------------
__global__ __launch_bounds__(256) void gemm_wmma_kernel(
    const _Float16* __restrict__ X, const _Float16* __restrict__ Wt,
    float* __restrict__ XP, int N) {
  __shared__ _Float16 As[128 * HF];   // 32 KB A panel

  const int tid   = threadIdx.x;
  const int wave  = tid >> 5;
  const int lane  = tid & 31;
  const int lhalf = lane & 15;
  const int kbase = (lane >> 4) * 8;   // 0 or 8 per ISA 16-bit A/B layout
  const int n0    = wave * 16;
  const int mbase = blockIdx.x * 128;

  // --- stage A panel into LDS: 2048 x 16B chunks, 8 per thread -------------
#pragma unroll
  for (int i = 0; i < 8; ++i) {
    int chunk = tid + i * 256;            // 0..2047
    int rowi  = chunk >> 4;               // 16 chunks (256B) per row
    int coff  = (chunk & 15) * 8;         // halves
    int gm    = mbase + rowi;
    int gmc   = (gm < N) ? gm : 0;        // clamp; padded rows never stored
    v4u d = *(const v4u*)(X + (size_t)gmc * HF + coff);
    *(v4u*)(As + rowi * HF + coff) = d;
  }

  // --- B fragments for all 4 K-chunks (overlap with LDS fill) --------------
  union { v16h h; v4u u[2]; } bfrag[4];
  const _Float16* wcol = Wt + (size_t)(n0 + lhalf) * HF;
#pragma unroll
  for (int kc = 0; kc < 4; ++kc) {
    int k0 = kc * 32;
    bfrag[kc].u[0] = *(const v4u*)(wcol + k0 + kbase);        // K = k0+kbase..+7
    bfrag[kc].u[1] = *(const v4u*)(wcol + k0 + 16 + kbase);   // K = k0+16+kbase..+7
  }

  __syncthreads();

  const bool fullblk = (mbase + 128 <= N);
  const int  rowoff  = (lane >> 4) * 8;
  const _Float16* arow = As + (size_t)lhalf * HF;   // local row lhalf of tile

#pragma unroll
  for (int rt = 0; rt < 8; ++rt) {
    int m0 = mbase + rt * 16;
    const _Float16* ap = arow + rt * 16 * HF;
    v8f c = {};
#pragma unroll
    for (int kc = 0; kc < 4; ++kc) {
      int k0 = kc * 32;
      union { v16h h; v4u u[2]; } a;
      a.u[0] = *(const v4u*)(ap + k0 + kbase);
      a.u[1] = *(const v4u*)(ap + k0 + 16 + kbase);
      c = __builtin_amdgcn_wmma_f32_16x16x32_f16(
              false, a.h, false, bfrag[kc].h, (short)0, c, false, false);
    }
    // C/D layout: VGPR r -> M = r (lanes 0-15) / 8+r (lanes 16-31), N = n0+lhalf
    if (fullblk) {
      float* base = XP + (size_t)(m0 + rowoff) * HF + n0 + lhalf;
#pragma unroll
      for (int r = 0; r < 8; ++r) base[(size_t)r * HF] = c[r];
    } else {
#pragma unroll
      for (int r = 0; r < 8; ++r) {
        int mm = m0 + rowoff + r;
        if (mm < N) XP[(size_t)mm * HF + n0 + lhalf] = c[r];
      }
    }
  }
}

// ---------------------------------------------------------------------------
// 5) Edge scatter: agg[col] += xp[row] * nrm.  One wave per edge, float4/lane.
// ---------------------------------------------------------------------------
__global__ __launch_bounds__(256) void edge_agg_kernel(
    const int* __restrict__ row, const int* __restrict__ col,
    const float* __restrict__ nrm, const float* __restrict__ xp,
    float* __restrict__ agg, int E) {
  int wid  = (blockIdx.x * blockDim.x + threadIdx.x) >> 5;
  int lane = threadIdx.x & 31;
  if (wid >= E) return;
  int r = row[wid], c = col[wid];
  float nv = nrm[wid];
  float4 v = ((const float4*)(xp + (size_t)r * HF))[lane];
  float* dst = agg + (size_t)c * HF + lane * 4;
  atomicAdd(dst + 0, v.x * nv);
  atomicAdd(dst + 1, v.y * nv);
  atomicAdd(dst + 2, v.z * nv);
  atomicAdd(dst + 3, v.w * nv);
}

// ---------------------------------------------------------------------------
// 6) Finalize: h = relu(agg + dinv^2 * xp + b) -> f16 (input to next GEMM)
// ---------------------------------------------------------------------------
__global__ __launch_bounds__(256) void finalize_kernel(
    const float* __restrict__ agg, const float* __restrict__ xp,
    const float* __restrict__ dinv, const float* __restrict__ b,
    _Float16* __restrict__ xh_out, int N) {
  int idx = blockIdx.x * blockDim.x + threadIdx.x;
  if (idx >= N * HF) return;
  int node = idx >> 7, f = idx & (HF - 1);
  float di = dinv[node];
  float v = agg[idx] + di * di * xp[idx] + b[f];
  xh_out[idx] = (_Float16)fmaxf(v, 0.0f);
}

// ---------------------------------------------------------------------------
// 7) Layer 2: xp2[i] = dot(h[i], W2).  One wave per node, wave reduce.
// ---------------------------------------------------------------------------
__global__ __launch_bounds__(256) void dot_kernel(
    const _Float16* __restrict__ xh, const float* __restrict__ W2,
    float* __restrict__ xp2, int N) {
  int wid  = (blockIdx.x * blockDim.x + threadIdx.x) >> 5;
  int lane = threadIdx.x & 31;
  if (wid >= N) return;
  const _Float16* p = xh + (size_t)wid * HF + lane * 4;
  const float*    w = W2 + lane * 4;
  float s = (float)p[0] * w[0] + (float)p[1] * w[1] +
            (float)p[2] * w[2] + (float)p[3] * w[3];
#pragma unroll
  for (int off = 16; off; off >>= 1) s += __shfl_xor(s, off, 32);
  if (lane == 0) xp2[wid] = s;
}

__global__ void edge_agg1_kernel(const int* __restrict__ row,
                                 const int* __restrict__ col,
                                 const float* __restrict__ nrm,
                                 const float* __restrict__ xp2,
                                 float* __restrict__ agg1, int E) {
  int e = blockIdx.x * blockDim.x + threadIdx.x;
  if (e < E) atomicAdd(&agg1[col[e]], xp2[row[e]] * nrm[e]);
}

__global__ void finalize1_kernel(const float* __restrict__ agg1,
                                 const float* __restrict__ xp2,
                                 const float* __restrict__ dinv,
                                 const float* __restrict__ b2,
                                 float* __restrict__ out, int N) {
  int i = blockIdx.x * blockDim.x + threadIdx.x;
  if (i >= N) return;
  float di = dinv[i];
  out[i] = agg1[i] + di * di * xp2[i] + b2[0];
}

// ---------------------------------------------------------------------------
// Host-side launch
// ---------------------------------------------------------------------------
static inline size_t align256(size_t x) { return (x + 255) & ~(size_t)255; }

extern "C" void kernel_launch(void* const* d_in, const int* in_sizes, int n_in,
                              void* d_out, int out_size, void* d_ws, size_t ws_size,
                              hipStream_t stream) {
  const float* feat  = (const float*)d_in[0];
  const int*   eidx  = (const int*)d_in[1];
  const float* ew    = (const float*)d_in[2];
  const float* W0    = (const float*)d_in[3];
  const float* b0    = (const float*)d_in[4];
  const float* W1    = (const float*)d_in[5];
  const float* b1    = (const float*)d_in[6];
  const float* W2    = (const float*)d_in[7];
  const float* b2    = (const float*)d_in[8];
  float* out = (float*)d_out;

  const int N = in_sizes[0] / HF;
  const int E = in_sizes[2];
  const int* erow = eidx;
  const int* ecol = eidx + E;

  // Workspace carve
  char* ws = (char*)d_ws;
  size_t off = 0;
  _Float16* xh  = (_Float16*)(ws + off); off += align256((size_t)N * HF * 2);
  float*    xp  = (float*)(ws + off);    off += align256((size_t)N * HF * 4);
  float*    agg = (float*)(ws + off);    off += align256((size_t)N * HF * 4);
  _Float16* Wt0 = (_Float16*)(ws + off); off += align256((size_t)HF * HF * 2);
  _Float16* Wt1 = (_Float16*)(ws + off); off += align256((size_t)HF * HF * 2);
  float*    nrm = (float*)(ws + off);    off += align256((size_t)E * 4);
  float*    deg = (float*)(ws + off);    off += align256((size_t)N * 4);
  float*    din = (float*)(ws + off);    off += align256((size_t)N * 4);
  float*    xp2 = (float*)(ws + off);    off += align256((size_t)N * 4);
  float*    ag1 = (float*)(ws + off);    off += align256((size_t)N * 4);
  (void)ws_size; (void)n_in; (void)out_size;

  const int TB = 256;
  dim3 blk(TB);
  int nWavesN   = (N * 32 + TB - 1) / TB;        // one wave per node
  int nWavesE   = (E * 32 + TB - 1) / TB;        // one wave per edge
  int nThrN     = (N + TB - 1) / TB;
  int nThrE     = (E + TB - 1) / TB;
  int nThrNF    = (N * HF + TB - 1) / TB;
  int nGemmBlks = (N + 127) / 128;

  // --- Preprocessing -------------------------------------------------------
  rownorm_kernel<<<nWavesN, blk, 0, stream>>>(feat, xh, N);
  deg_init_kernel<<<nThrN, blk, 0, stream>>>(deg, N);
  deg_accum_kernel<<<nThrE, blk, 0, stream>>>(ecol, ew, deg, E);
  dinv_kernel<<<nThrN, blk, 0, stream>>>(deg, din, N);
  edgenorm_kernel<<<nThrE, blk, 0, stream>>>(erow, ecol, ew, din, nrm, E);
  transpose_w_kernel<<<HF, HF, 0, stream>>>(W0, Wt0);
  transpose_w_kernel<<<HF, HF, 0, stream>>>(W1, Wt1);

  // --- Layer 0 -------------------------------------------------------------
  gemm_wmma_kernel<<<nGemmBlks, blk, 0, stream>>>(xh, Wt0, xp, N);
  hipMemsetAsync(agg, 0, (size_t)N * HF * 4, stream);
  edge_agg_kernel<<<nWavesE, blk, 0, stream>>>(erow, ecol, nrm, xp, agg, E);
  finalize_kernel<<<nThrNF, blk, 0, stream>>>(agg, xp, din, b0, xh, N);

  // --- Layer 1 -------------------------------------------------------------
  gemm_wmma_kernel<<<nGemmBlks, blk, 0, stream>>>(xh, Wt1, xp, N);
  hipMemsetAsync(agg, 0, (size_t)N * HF * 4, stream);
  edge_agg_kernel<<<nWavesE, blk, 0, stream>>>(erow, ecol, nrm, xp, agg, E);
  finalize_kernel<<<nThrNF, blk, 0, stream>>>(agg, xp, din, b1, xh, N);

  // --- Layer 2 (H -> 1) ----------------------------------------------------
  dot_kernel<<<nWavesN, blk, 0, stream>>>(xh, W2, xp2, N);
  hipMemsetAsync(ag1, 0, (size_t)N * 4, stream);
  edge_agg1_kernel<<<nThrE, blk, 0, stream>>>(erow, ecol, nrm, xp2, ag1, E);
  finalize1_kernel<<<nThrN, blk, 0, stream>>>(ag1, xp2, din, b2, out, N);
}
